// Linear_Transformer_14714557956574
// MI455X (gfx1250) — matrix operand order repaired and checked by reading, not compiled, not run on hardware
//
#include <hip/hip_runtime.h>

typedef __attribute__((ext_vector_type(16))) __bf16 v16bf;
typedef __attribute__((ext_vector_type(8)))  __bf16 v8bf;
typedef __attribute__((ext_vector_type(8)))  float  v8f;

static constexpr int Bsz = 64, Sseq = 512, Dm = 512, BS = Bsz * Sseq, FFdim = 2048;

__device__ __forceinline__ __bf16 f2bf(float f) {
  unsigned u = __builtin_bit_cast(unsigned, f);
  unsigned r = u + 0x7FFFu + ((u >> 16) & 1u);
  unsigned short s = (unsigned short)(r >> 16);
  return __builtin_bit_cast(__bf16, s);
}
__device__ __forceinline__ float bf2f(__bf16 b) {
  unsigned short s = __builtin_bit_cast(unsigned short, b);
  return __builtin_bit_cast(float, ((unsigned)s) << 16);
}

// ---------------------------------------------------------------------------
// Weight convert + transpose: W[K,N] fp32 -> WT[N,K] bf16
// ---------------------------------------------------------------------------
__global__ void cvt_transpose(const float* __restrict__ W, __bf16* __restrict__ WT,
                              int K, int N) {
  int idx = blockIdx.x * 256 + threadIdx.x;
  if (idx >= N * K) return;
  int n = idx / K, k = idx % K;
  WT[idx] = f2bf(W[(size_t)k * N + n]);
}

// ---------------------------------------------------------------------------
// LayerNorm over D=512, output bf16. grid = rows, block = 256
// ---------------------------------------------------------------------------
__global__ void ln_kernel(const float* __restrict__ X, const float* __restrict__ g,
                          const float* __restrict__ beta, __bf16* __restrict__ Y) {
  int row = blockIdx.x, tid = threadIdx.x;
  const float* xr = X + (size_t)row * Dm;
  float v0 = xr[tid], v1 = xr[tid + 256];
  __shared__ float red[256];
  red[tid] = v0 + v1;
  __syncthreads();
  for (int o = 128; o > 0; o >>= 1) {
    if (tid < o) red[tid] += red[tid + o];
    __syncthreads();
  }
  float mu = red[0] * (1.f / 512.f);
  __syncthreads();
  float d0 = v0 - mu, d1 = v1 - mu;
  red[tid] = d0 * d0 + d1 * d1;
  __syncthreads();
  for (int o = 128; o > 0; o >>= 1) {
    if (tid < o) red[tid] += red[tid + o];
    __syncthreads();
  }
  float rstd = rsqrtf(red[0] * (1.f / 512.f) + 1e-5f);
  __bf16* yr = Y + (size_t)row * Dm;
  yr[tid]       = f2bf(d0 * rstd * g[tid] + beta[tid]);
  yr[tid + 256] = f2bf(d1 * rstd * g[tid + 256] + beta[tid + 256]);
}

// ---------------------------------------------------------------------------
// Generic bf16 WMMA GEMM: C[M,N] = A[M,K] * BT[N,K]^T (+ epilogue)
// EPI: 0 = store bf16 (no bias)
//      1 = fp32 out = res + acc + bias
//      2 = bf16 out = gelu(acc + bias)        (exact erf gelu)
//      3 = bf16 out = tanh(acc + bias)
//      4 = fp32 out = acc + bias
// Wave tile = 64x64: 4 A frags x 4 B frags x 16 f32 accumulators.
// 16 WMMAs per 8 fragment loads per k-step (2x the intensity of 32x64).
// ---------------------------------------------------------------------------
template <int EPI>
__global__ void __launch_bounds__(256) gemm_wmma(
    const __bf16* __restrict__ A, const __bf16* __restrict__ BT,
    const float* __restrict__ bias, const float* __restrict__ res,
    float* __restrict__ outF, __bf16* __restrict__ outB,
    int M, int N, int K) {
  int lane = threadIdx.x & 31;
  int wave = blockIdx.x * 8 + (threadIdx.x >> 5);
  int tiles_m = M >> 6, tiles_n = N >> 6;
  if (wave >= tiles_m * tiles_n) return;
  int tm = (wave % tiles_m) << 6;
  int tn = (wave / tiles_m) << 6;
  int r = lane & 15, half = lane >> 4;

  const __bf16* ap[4];
#pragma unroll
  for (int i = 0; i < 4; i++) ap[i] = A + (size_t)(tm + 16 * i + r) * K + 16 * half;
  const __bf16* bp[4];
#pragma unroll
  for (int j = 0; j < 4; j++) bp[j] = BT + (size_t)(tn + 16 * j + r) * K + 16 * half;

  v8f acc[4][4] = {};
  for (int k = 0; k < K; k += 32) {
    v16bf af[4];
#pragma unroll
    for (int i = 0; i < 4; i++) {
      v8bf lo = *(const v8bf*)(ap[i] + k);
      v8bf hi = *(const v8bf*)(ap[i] + k + 8);
#pragma unroll
      for (int u = 0; u < 8; u++) { af[i][u] = lo[u]; af[i][u + 8] = hi[u]; }
    }
    if (k + 32 < K) {
#pragma unroll
      for (int i = 0; i < 4; i++)
        __builtin_prefetch((const void*)(ap[i] + k + 32), 0, 3);
#pragma unroll
      for (int j = 0; j < 4; j++)
        __builtin_prefetch((const void*)(bp[j] + k + 32), 0, 3);
    }
#pragma unroll
    for (int j = 0; j < 4; j++) {
      v16bf bf = *(const v16bf*)(bp[j] + k);
      acc[0][j] = __builtin_amdgcn_wmma_f32_16x16x32_bf16(false, af[0], false, bf, (short)0, acc[0][j], false, false);
      acc[1][j] = __builtin_amdgcn_wmma_f32_16x16x32_bf16(false, af[1], false, bf, (short)0, acc[1][j], false, false);
      acc[2][j] = __builtin_amdgcn_wmma_f32_16x16x32_bf16(false, af[2], false, bf, (short)0, acc[2][j], false, false);
      acc[3][j] = __builtin_amdgcn_wmma_f32_16x16x32_bf16(false, af[3], false, bf, (short)0, acc[3][j], false, false);
    }
  }

#pragma unroll
  for (int i = 0; i < 4; i++) {
#pragma unroll
    for (int j = 0; j < 4; j++) {
      int col = tn + 16 * j + r;
      float bv = (EPI != 0) ? bias[col] : 0.f;
#pragma unroll
      for (int e = 0; e < 8; e++) {
        int row = tm + 16 * i + 8 * half + e;
        size_t idx = (size_t)row * N + col;
        float v = acc[i][j][e] + bv;
        if (EPI == 0) {
          outB[idx] = f2bf(v);
        } else if (EPI == 1) {
          outF[idx] = res[idx] + v;
        } else if (EPI == 2) {
          outB[idx] = f2bf(0.5f * v * (1.f + erff(v * 0.70710678118f)));
        } else if (EPI == 3) {
          outB[idx] = f2bf(tanhf(v));
        } else {
          outF[idx] = v;
        }
      }
    }
  }
}

// ---------------------------------------------------------------------------
// Local window attention, heads 0..1, window=128, look back/fwd 1.
// grid = B*2*4 blocks, block = 128 threads (1 thread = 1 query).
// Keys processed in two 192-wide chunks staged in LDS (48KB), online softmax.
// ---------------------------------------------------------------------------
__global__ void __launch_bounds__(128) local_attn(
    const __bf16* __restrict__ Q, const __bf16* __restrict__ Kb,
    const __bf16* __restrict__ Vb, __bf16* __restrict__ A) {
  int blk = blockIdx.x;
  int b = blk >> 3;
  int head = (blk >> 2) & 1;
  int iw = blk & 3;
  int base = head * 64;

  __shared__ __bf16 kc[192 * 64];
  __shared__ __bf16 vc[192 * 64];

  int qi = iw * 128 + threadIdx.x;
  float q[64];
  {
    const __bf16* qp = Q + ((size_t)(b * Sseq + qi)) * Dm + base;
    for (int d = 0; d < 64; d++) q[d] = bf2f(qp[d]);
  }
  float m = -1e30f, l = 0.f, o[64];
  for (int e = 0; e < 64; e++) o[e] = 0.f;

  for (int c = 0; c < 2; c++) {
    __syncthreads();
    for (int i = threadIdx.x; i < 192 * 8; i += 128) {
      int j = i >> 3, d8 = (i & 7) * 8;
      int pos = (iw - 1) * 128 + c * 192 + j;
      if (pos >= 0 && pos < Sseq) {
        *(v8bf*)&kc[j * 64 + d8] =
            *(const v8bf*)(Kb + ((size_t)(b * Sseq + pos)) * Dm + base + d8);
        *(v8bf*)&vc[j * 64 + d8] =
            *(const v8bf*)(Vb + ((size_t)(b * Sseq + pos)) * Dm + base + d8);
      }
    }
    __syncthreads();
    for (int j = 0; j < 192; j++) {
      int pos = (iw - 1) * 128 + c * 192 + j;
      if (pos < 0 || pos >= Sseq) continue;
      float s = 0.f;
      for (int d = 0; d < 64; d++) s += q[d] * bf2f(kc[j * 64 + d]);
      s *= 0.125f;
      float mn = fmaxf(m, s);
      float corr = __expf(m - mn);
      float p = __expf(s - mn);
      l = l * corr + p;
      for (int e = 0; e < 64; e++) o[e] = o[e] * corr + p * bf2f(vc[j * 64 + e]);
      m = mn;
    }
  }
  float inv = 1.f / l;
  __bf16* op = A + ((size_t)(b * Sseq + qi)) * Dm + base;
  for (int e = 0; e < 64; e++) op[e] = f2bf(o[e] * inv);
}

// ---------------------------------------------------------------------------
// Linear attention, heads 2..7: ctx[d,e] = sum_n softmax_n(k)[n,d] * v[n,e]
// grid = B*6 blocks, block = 256.
// ---------------------------------------------------------------------------
__global__ void __launch_bounds__(256) lin_ctx(
    const __bf16* __restrict__ Kb, const __bf16* __restrict__ Vb,
    float* __restrict__ ctx) {
  int blk = blockIdx.x;
  int b = blk / 6;
  int hh = blk % 6;          // 0..5
  int base = (hh + 2) * 64;  // heads 2..7

  __shared__ float colmax[64], colinv[64];
  __shared__ __bf16 kc[64 * 64];
  __shared__ __bf16 vc[64 * 64];

  int t = threadIdx.x;
  if (t < 64) {
    int d = t;
    float mx = -1e30f;
    for (int n = 0; n < Sseq; n++)
      mx = fmaxf(mx, bf2f(Kb[((size_t)(b * Sseq + n)) * Dm + base + d]));
    float s = 0.f;
    for (int n = 0; n < Sseq; n++)
      s += __expf(bf2f(Kb[((size_t)(b * Sseq + n)) * Dm + base + d]) - mx);
    colmax[d] = mx;
    colinv[d] = 1.f / s;
  }
  __syncthreads();

  int d = t >> 2;
  int e0 = (t & 3) * 16;
  float accv[16];
  for (int u = 0; u < 16; u++) accv[u] = 0.f;

  for (int n0 = 0; n0 < Sseq; n0 += 64) {
    // stage 64 rows of k and v (16 contiguous bf16 per thread each)
    int nr = t >> 2, cs = (t & 3) * 16;
    const __bf16* kp = Kb + ((size_t)(b * Sseq + n0 + nr)) * Dm + base + cs;
    const __bf16* vp = Vb + ((size_t)(b * Sseq + n0 + nr)) * Dm + base + cs;
    *(v8bf*)&kc[nr * 64 + cs]     = *(const v8bf*)kp;
    *(v8bf*)&kc[nr * 64 + cs + 8] = *(const v8bf*)(kp + 8);
    *(v8bf*)&vc[nr * 64 + cs]     = *(const v8bf*)vp;
    *(v8bf*)&vc[nr * 64 + cs + 8] = *(const v8bf*)(vp + 8);
    __syncthreads();
    for (int nl = 0; nl < 64; nl++) {
      float p = __expf(bf2f(kc[nl * 64 + d]) - colmax[d]) * colinv[d];
      for (int u = 0; u < 16; u++) accv[u] += p * bf2f(vc[nl * 64 + e0 + u]);
    }
    __syncthreads();
  }
  float* cp = ctx + ((size_t)blk) * 64 * 64 + d * 64 + e0;
  for (int u = 0; u < 16; u++) cp[u] = accv[u];
}

// out[n,e] = sum_d (softmax_d(q)[n,d] * dh^-0.5) * ctx[d,e]
__global__ void __launch_bounds__(256) lin_out(
    const __bf16* __restrict__ Q, const float* __restrict__ ctx,
    __bf16* __restrict__ A) {
  int blk = blockIdx.x;
  int b = blk / 6;
  int hh = blk % 6;
  int base = (hh + 2) * 64;

  __shared__ float cs[64 * 64];
  {
    const float* cp = ctx + ((size_t)blk) * 64 * 64;
    for (int u = 0; u < 16; u++) cs[threadIdx.x * 16 + u] = cp[threadIdx.x * 16 + u];
  }
  __syncthreads();

  for (int pass = 0; pass < 2; pass++) {
    int qn = threadIdx.x + pass * 256;
    const __bf16* qp = Q + ((size_t)(b * Sseq + qn)) * Dm + base;
    float f[64], mx = -1e30f;
    for (int d = 0; d < 64; d++) { f[d] = bf2f(qp[d]); mx = fmaxf(mx, f[d]); }
    float s = 0.f;
    for (int d = 0; d < 64; d++) { f[d] = __expf(f[d] - mx); s += f[d]; }
    float scale = 0.125f / s;  // dh^-0.5 = 1/8
    __bf16* op = A + ((size_t)(b * Sseq + qn)) * Dm + base;
    for (int e = 0; e < 64; e++) {
      float acc = 0.f;
      for (int d = 0; d < 64; d++) acc += f[d] * cs[d * 64 + e];
      op[e] = f2bf(acc * scale);
    }
  }
}

// latent[b,d] = mean over s of h[b,s,d]; output bf16. grid = B, block = 256
__global__ void mean_kernel(const float* __restrict__ H, __bf16* __restrict__ L) {
  int b = blockIdx.x;
  for (int d = threadIdx.x; d < Dm; d += 256) {
    float s = 0.f;
    for (int n = 0; n < Sseq; n++) s += H[((size_t)(b * Sseq + n)) * Dm + d];
    L[(size_t)b * Dm + d] = f2bf(s * (1.f / 512.f));
  }
}

// ---------------------------------------------------------------------------
// Host launcher
// ---------------------------------------------------------------------------
static inline size_t al256(size_t x) { return (x + 255) & ~(size_t)255; }

extern "C" void kernel_launch(void* const* d_in, const int* in_sizes, int n_in,
                              void* d_out, int out_size, void* d_ws, size_t ws_size,
                              hipStream_t stream) {
  const float* x    = (const float*)d_in[0];
  const float* Wq   = (const float*)d_in[1];
  const float* Wk   = (const float*)d_in[2];
  const float* Wv   = (const float*)d_in[3];
  const float* Wo   = (const float*)d_in[4];
  const float* bo   = (const float*)d_in[5];
  const float* ln1g = (const float*)d_in[6];
  const float* ln1b = (const float*)d_in[7];
  const float* Wf1  = (const float*)d_in[8];
  const float* bf1  = (const float*)d_in[9];
  const float* Wf2  = (const float*)d_in[10];
  const float* bf2  = (const float*)d_in[11];
  const float* ln2g = (const float*)d_in[12];
  const float* ln2b = (const float*)d_in[13];
  const float* fc1w = (const float*)d_in[14];
  const float* fc1b = (const float*)d_in[15];
  const float* fc2w = (const float*)d_in[16];
  const float* fc2b = (const float*)d_in[17];
  float* outp = (float*)d_out;

  char* p = (char*)d_ws;
  auto alloc = [&](size_t bytes) { void* r = (void*)p; p += al256(bytes); return r; };

  float*  hbuf = (float*)alloc((size_t)BS * Dm * 4);
  __bf16* ybuf = (__bf16*)alloc((size_t)BS * Dm * 2);
  __bf16* qbuf = (__bf16*)alloc((size_t)BS * Dm * 2);
  __bf16* kbuf = (__bf16*)alloc((size_t)BS * Dm * 2);
  __bf16* vbuf = (__bf16*)alloc((size_t)BS * Dm * 2);
  __bf16* abuf = (__bf16*)alloc((size_t)BS * Dm * 2);
  __bf16* gbuf = (__bf16*)alloc((size_t)BS * FFdim * 2);
  float*  ctxb = (float*)alloc((size_t)Bsz * 6 * 64 * 64 * 4);
  __bf16* lat  = (__bf16*)alloc((size_t)Bsz * Dm * 2);
  __bf16* t1   = (__bf16*)alloc((size_t)Bsz * Dm * 2);
  __bf16* wqT  = (__bf16*)alloc((size_t)2 * Dm * Dm * 2);
  __bf16* wkT  = (__bf16*)alloc((size_t)2 * Dm * Dm * 2);
  __bf16* wvT  = (__bf16*)alloc((size_t)2 * Dm * Dm * 2);
  __bf16* woT  = (__bf16*)alloc((size_t)2 * Dm * Dm * 2);
  __bf16* wf1T = (__bf16*)alloc((size_t)2 * Dm * FFdim * 2);
  __bf16* wf2T = (__bf16*)alloc((size_t)2 * Dm * FFdim * 2);
  __bf16* fc1T = (__bf16*)alloc((size_t)Dm * Dm * 2);
  __bf16* fc2T = (__bf16*)alloc((size_t)128 * Dm * 2);

  // h = x
  hipMemcpyAsync(hbuf, x, (size_t)BS * Dm * 4, hipMemcpyDeviceToDevice, stream);

  auto cvt = [&](const float* W, __bf16* WT, int K, int N) {
    int elems = N * K;
    cvt_transpose<<<(elems + 255) / 256, 256, 0, stream>>>(W, WT, K, N);
  };
  for (int l = 0; l < 2; l++) {
    cvt(Wq + (size_t)l * Dm * Dm, wqT + (size_t)l * Dm * Dm, Dm, Dm);
    cvt(Wk + (size_t)l * Dm * Dm, wkT + (size_t)l * Dm * Dm, Dm, Dm);
    cvt(Wv + (size_t)l * Dm * Dm, wvT + (size_t)l * Dm * Dm, Dm, Dm);
    cvt(Wo + (size_t)l * Dm * Dm, woT + (size_t)l * Dm * Dm, Dm, Dm);
    cvt(Wf1 + (size_t)l * Dm * FFdim, wf1T + (size_t)l * Dm * FFdim, Dm, FFdim);
    cvt(Wf2 + (size_t)l * Dm * FFdim, wf2T + (size_t)l * Dm * FFdim, FFdim, Dm);
  }
  cvt(fc1w, fc1T, Dm, Dm);
  cvt(fc2w, fc2T, Dm, 128);

  auto gemm_blocks = [](int M, int N) { return (((M >> 6) * (N >> 6)) + 7) >> 3; };

  for (int l = 0; l < 2; l++) {
    const __bf16* wq = wqT + (size_t)l * Dm * Dm;
    const __bf16* wk = wkT + (size_t)l * Dm * Dm;
    const __bf16* wv = wvT + (size_t)l * Dm * Dm;
    const __bf16* wo = woT + (size_t)l * Dm * Dm;
    const __bf16* w1 = wf1T + (size_t)l * Dm * FFdim;
    const __bf16* w2 = wf2T + (size_t)l * Dm * FFdim;

    // y = LN1(h)
    ln_kernel<<<BS, 256, 0, stream>>>(hbuf, ln1g + l * Dm, ln1b + l * Dm, ybuf);
    // q,k,v = y @ W{q,k,v}
    gemm_wmma<0><<<gemm_blocks(BS, Dm), 256, 0, stream>>>(ybuf, wq, nullptr, nullptr, nullptr, qbuf, BS, Dm, Dm);
    gemm_wmma<0><<<gemm_blocks(BS, Dm), 256, 0, stream>>>(ybuf, wk, nullptr, nullptr, nullptr, kbuf, BS, Dm, Dm);
    gemm_wmma<0><<<gemm_blocks(BS, Dm), 256, 0, stream>>>(ybuf, wv, nullptr, nullptr, nullptr, vbuf, BS, Dm, Dm);
    // attention -> abuf
    local_attn<<<Bsz * 2 * 4, 128, 0, stream>>>(qbuf, kbuf, vbuf, abuf);
    lin_ctx<<<Bsz * 6, 256, 0, stream>>>(kbuf, vbuf, ctxb);
    lin_out<<<Bsz * 6, 256, 0, stream>>>(qbuf, ctxb, abuf);
    // h = h + a @ Wo + bo
    gemm_wmma<1><<<gemm_blocks(BS, Dm), 256, 0, stream>>>(abuf, wo, bo + l * Dm, hbuf, hbuf, nullptr, BS, Dm, Dm);
    // y = LN2(h)
    ln_kernel<<<BS, 256, 0, stream>>>(hbuf, ln2g + l * Dm, ln2b + l * Dm, ybuf);
    // g = gelu(y @ Wf1 + bf1)
    gemm_wmma<2><<<gemm_blocks(BS, FFdim), 256, 0, stream>>>(ybuf, w1, bf1 + l * FFdim, nullptr, nullptr, gbuf, BS, FFdim, Dm);
    // h = h + g @ Wf2 + bf2
    gemm_wmma<1><<<gemm_blocks(BS, Dm), 256, 0, stream>>>(gbuf, w2, bf2 + l * Dm, hbuf, hbuf, nullptr, BS, Dm, FFdim);
  }

  // head
  mean_kernel<<<Bsz, 256, 0, stream>>>(hbuf, lat);
  gemm_wmma<3><<<gemm_blocks(Bsz, Dm), 256, 0, stream>>>(lat, fc1T, fc1b, nullptr, nullptr, t1, Bsz, Dm, Dm);
  gemm_wmma<4><<<gemm_blocks(Bsz, 128), 256, 0, stream>>>(t1, fc2T, fc2b, nullptr, outp, nullptr, Bsz, 128, Dm);
}
